// GraphSAGE_50663434223943
// MI455X (gfx1250) — compile-verified, matrix operand order checked by reading, not run on hardware
//
#include <hip/hip_runtime.h>
#include <hip/hip_bf16.h>

typedef __attribute__((ext_vector_type(16))) _Float16 v16h;
typedef __attribute__((ext_vector_type(8)))  float    v8f;

#define N_NODES  262144
#define N_EDGES  1048576
#define N_GRAPHS 8192
#define IN_CH    64
#define HID      256
#define SLOPE    0.1f

#if __has_builtin(__builtin_amdgcn_sched_barrier)
#define SCHED_FENCE() __builtin_amdgcn_sched_barrier(0)
#else
#define SCHED_FENCE()
#endif

// ---------------------------------------------------------------------------
// Pack a row-major f32 weight matrix W[K][N] into per-lane v16h B-fragments
// matching the CDNA5 16-bit B operand layout for V_WMMA_F32_16X16X32_F16.
// Fragment index: (kt * (N/16) + nt) * 32 + lane ; lane holds 16 halves:
//   n  = nt*16 + (lane & 15), khalf = lane >> 4
//   element e -> k = kt*32 + khalf*8 + (e < 8 ? e : e + 8)
// ---------------------------------------------------------------------------
__global__ __launch_bounds__(256)
void pack_weights_kernel(const float* __restrict__ W, v16h* __restrict__ out,
                         int K, int N) {
    int idx   = blockIdx.x * blockDim.x + threadIdx.x;
    int NT    = N >> 4;
    int total = (K >> 5) * NT * 32;
    if (idx >= total) return;
    int lane = idx & 31;
    int nt   = (idx >> 5) % NT;
    int kt   = (idx >> 5) / NT;
    int n    = nt * 16 + (lane & 15);
    int kh   = lane >> 4;
    v16h v;
#pragma unroll
    for (int e = 0; e < 16; ++e) {
        int k = kt * 32 + kh * 8 + (e < 8 ? e : e + 8);
        v[e] = (_Float16)W[(size_t)k * N + n];
    }
    out[idx] = v;
}

// ---------------------------------------------------------------------------
// Fused WMMA GEMM: C[M][256] = A1[M][K] (*scale1 per row) @ W1 + bias
//                              (+ A2[M][K] @ W2)   ; optional LeakyReLU.
// Block: 256 threads = 8 waves; block tile 64(M) x 256(N);
// wave tile 16(M) x 128(N) = 8 WMMA accumulator tiles.
// Single static inner-burst site (runtime op-loop over the 1-2 operands):
// issue raw-A loads + all 8 B-fragment loads, then a sched_barrier pins the
// batch; the cvt + 8-WMMA burst after it drains the loads with staggered
// waitcnts instead of per-tile wait-0 serialization.
// ---------------------------------------------------------------------------
__global__ __launch_bounds__(256)
void gemm_wmma_kernel(const float* __restrict__ A1,
                      const float* __restrict__ scale1,   // per-row scale or null
                      const v16h*  __restrict__ W1,
                      const float* __restrict__ A2,       // second operand or null
                      const v16h*  __restrict__ W2,
                      const float* __restrict__ bias,
                      float*       __restrict__ C,
                      int M, int K, int relu) {
    const int lane  = threadIdx.x & 31;
    const int w     = threadIdx.x >> 5;
    const int wm    = w & 3;            // wave M-row 0..3
    const int wn    = w >> 2;           // wave N-col 0..1 (128 cols each)
    const int m0    = blockIdx.x * 64 + wm * 16;
    const int lrow  = lane & 15;
    const int khalf = lane >> 4;
    const int NT    = 16;               // N/16 with N = 256

    // init accumulators with broadcast bias
    v8f acc[8];
#pragma unroll
    for (int t = 0; t < 8; ++t) {
        float bv = bias[((wn << 3) + t) * 16 + lrow];
#pragma unroll
        for (int r = 0; r < 8; ++r) acc[t][r] = bv;
    }

    const int    row  = m0 + lrow;
    const float  s1   = scale1 ? scale1[row] : 1.0f;
    const float* a1p  = A1 + (size_t)row * K + khalf * 8;
    const float* a2p  = A2 ? (A2 + (size_t)row * K + khalf * 8) : nullptr;
    const int    nops = a2p ? 2 : 1;

    const int ktiles = K >> 5;
    for (int kt = 0; kt < ktiles; ++kt) {
        for (int op = 0; op < nops; ++op) {
            const float* apq = (op == 0) ? a1p : a2p;
            const v16h*  wq  = (op == 0) ? W1 : W2;
            const float  s   = (op == 0) ? s1 : 1.0f;

            // ---- issue raw A chunk + all 8 B fragments as one load batch ----
            const float4* pa = (const float4*)(apq + kt * 32);
            float4 t0 = pa[0], t1 = pa[1], t2 = pa[4], t3 = pa[5];
            v16h b[8];
#pragma unroll
            for (int t = 0; t < 8; ++t)
                b[t] = wq[(size_t)(kt * NT + (wn << 3) + t) * 32 + lane];

            // pin the batch: loads stay above, math stays below
            SCHED_FENCE();

            float ra[16] = {t0.x, t0.y, t0.z, t0.w, t1.x, t1.y, t1.z, t1.w,
                            t2.x, t2.y, t2.z, t2.w, t3.x, t3.y, t3.z, t3.w};
            v16h a;
#pragma unroll
            for (int e = 0; e < 16; ++e) a[e] = (_Float16)(ra[e] * s);
#pragma unroll
            for (int t = 0; t < 8; ++t)
                acc[t] = __builtin_amdgcn_wmma_f32_16x16x32_f16(
                    false, a, false, b[t], (short)0, acc[t], false, false);
        }
        if (kt + 1 < ktiles) {
            __builtin_prefetch(a1p + (kt + 1) * 32, 0, 3);
            if (a2p) __builtin_prefetch(a2p + (kt + 1) * 32, 0, 3);
        }
    }

    // epilogue: C/D layout -> row = m0 + khalf*8 + r, col = n0 + (lane&15)
    const int rbase = m0 + (khalf << 3);
#pragma unroll
    for (int t = 0; t < 8; ++t) {
        int n = ((wn << 3) + t) * 16 + lrow;
#pragma unroll
        for (int r = 0; r < 8; ++r) {
            float v = acc[t][r];
            if (relu) v = v > 0.0f ? v : SLOPE * v;
            C[(size_t)(rbase + r) * HID + n] = v;
        }
    }
}

// ---------------------------------------------------------------------------
// Degree accumulation: one thread per edge.
// ---------------------------------------------------------------------------
__global__ __launch_bounds__(256)
void degree_kernel(const int* __restrict__ dst, float* __restrict__ deg, int nEdges) {
    int e = blockIdx.x * blockDim.x + threadIdx.x;
    if (e >= nEdges) return;
    atomicAdd(&deg[dst[e]], 1.0f);
}

__global__ __launch_bounds__(256)
void invert_deg_kernel(float* __restrict__ deg, int n) {
    int i = blockIdx.x * blockDim.x + threadIdx.x;
    if (i >= n) return;
    float d = deg[i];
    deg[i] = 1.0f / fmaxf(d, 1.0f);
}

// ---------------------------------------------------------------------------
// Edge scatter-add: wave per edge, 8 channels per lane (256 ch).
// ---------------------------------------------------------------------------
__global__ __launch_bounds__(256)
void scatter_add_kernel(const int* __restrict__ src, const int* __restrict__ dst,
                        const float* __restrict__ h, float* __restrict__ agg,
                        int nEdges) {
    int e    = (blockIdx.x * blockDim.x + threadIdx.x) >> 5;
    int lane = threadIdx.x & 31;
    if (e >= nEdges) return;
    int s = src[e];
    int d = dst[e];
    const float4* hp = (const float4*)(h + (size_t)s * HID + lane * 8);
    float4 v0 = hp[0];
    float4 v1 = hp[1];
    float* ap = agg + (size_t)d * HID + lane * 8;
    atomicAdd(ap + 0, v0.x); atomicAdd(ap + 1, v0.y);
    atomicAdd(ap + 2, v0.z); atomicAdd(ap + 3, v0.w);
    atomicAdd(ap + 4, v1.x); atomicAdd(ap + 5, v1.y);
    atomicAdd(ap + 6, v1.z); atomicAdd(ap + 7, v1.w);
}

// ---------------------------------------------------------------------------
// Global add pool: wave per node, 8 channels per lane.
// ---------------------------------------------------------------------------
__global__ __launch_bounds__(256)
void pool_kernel(const int* __restrict__ batch, const float* __restrict__ h,
                 float* __restrict__ pooled, int nNodes) {
    int nidx = (blockIdx.x * blockDim.x + threadIdx.x) >> 5;
    int lane = threadIdx.x & 31;
    if (nidx >= nNodes) return;
    int g = batch[nidx];
    const float4* hp = (const float4*)(h + (size_t)nidx * HID + lane * 8);
    float4 v0 = hp[0];
    float4 v1 = hp[1];
    float* pp = pooled + (size_t)g * HID + lane * 8;
    atomicAdd(pp + 0, v0.x); atomicAdd(pp + 1, v0.y);
    atomicAdd(pp + 2, v0.z); atomicAdd(pp + 3, v0.w);
    atomicAdd(pp + 4, v1.x); atomicAdd(pp + 5, v1.y);
    atomicAdd(pp + 6, v1.z); atomicAdd(pp + 7, v1.w);
}

// ---------------------------------------------------------------------------
// Final projection: out[g] = hidden[g][:] . w2[:] + b2 ; wave per graph.
// ---------------------------------------------------------------------------
__global__ __launch_bounds__(256)
void out_kernel(const float* __restrict__ hidden, const float* __restrict__ w2,
                const float* __restrict__ b2, float* __restrict__ out, int G) {
    int g    = (blockIdx.x * blockDim.x + threadIdx.x) >> 5;
    int lane = threadIdx.x & 31;
    if (g >= G) return;
    const float* hp = hidden + (size_t)g * HID + lane * 8;
    const float* wp = w2 + lane * 8;
    float sum = 0.0f;
#pragma unroll
    for (int j = 0; j < 8; ++j) sum += hp[j] * wp[j];
#pragma unroll
    for (int off = 16; off > 0; off >>= 1) sum += __shfl_xor(sum, off, 32);
    if (lane == 0) out[g] = sum + b2[0];
}

// ---------------------------------------------------------------------------
extern "C" void kernel_launch(void* const* d_in, const int* in_sizes, int n_in,
                              void* d_out, int out_size, void* d_ws, size_t ws_size,
                              hipStream_t stream) {
    const float* x        = (const float*)d_in[0];
    const int*   ei       = (const int*)d_in[1];          // [2, N_EDGES]
    const int*   batch    = (const int*)d_in[2];
    const float* embed_w  = (const float*)d_in[3];        // [64,256]
    const float* embed_b  = (const float*)d_in[4];
    const float* lin_l_w  = (const float*)d_in[5];        // [3,256,256]
    const float* lin_l_b  = (const float*)d_in[6];        // [3,256]
    const float* lin_r_w  = (const float*)d_in[7];        // [3,256,256]
    const float* lin1_w   = (const float*)d_in[8];        // [256,256]
    const float* lin1_b   = (const float*)d_in[9];
    const float* lin2_w   = (const float*)d_in[10];       // [256,1]
    const float* lin2_b   = (const float*)d_in[11];
    float*       out      = (float*)d_out;

    const int* src = ei;
    const int* dst = ei + N_EDGES;

    // workspace layout
    char* base = (char*)d_ws;
    const size_t szH = (size_t)N_NODES * HID * sizeof(float);   // 268 MB
    float* hA     = (float*)(base);
    float* hB     = (float*)(base + szH);
    float* agg    = (float*)(base + 2 * szH);
    float* invdeg = (float*)(base + 3 * szH);
    float* pooled = (float*)(base + 3 * szH + (size_t)N_NODES * sizeof(float));
    float* hidden = (float*)((char*)pooled + (size_t)N_GRAPHS * HID * sizeof(float));
    v16h*  pack   = (v16h*)((char*)hidden + (size_t)N_GRAPHS * HID * sizeof(float));

    // packed-weight sub-regions (in v16h units)
    const size_t embedFrags = (IN_CH / 32) * (HID / 16) * 32;   // 1024
    const size_t matFrags   = (HID / 32) * (HID / 16) * 32;     // 4096
    v16h* pEmbed = pack;
    v16h* pL[3], *pR[3];
    for (int i = 0; i < 3; ++i) {
        pL[i] = pack + embedFrags + (size_t)(2 * i) * matFrags;
        pR[i] = pack + embedFrags + (size_t)(2 * i + 1) * matFrags;
    }
    v16h* pLin1 = pack + embedFrags + 6 * matFrags;

    // ---- pack weights into WMMA B-fragment layout ----
    pack_weights_kernel<<<(embedFrags + 255) / 256, 256, 0, stream>>>(embed_w, pEmbed, IN_CH, HID);
    for (int i = 0; i < 3; ++i) {
        pack_weights_kernel<<<(matFrags + 255) / 256, 256, 0, stream>>>(
            lin_l_w + (size_t)i * HID * HID, pL[i], HID, HID);
        pack_weights_kernel<<<(matFrags + 255) / 256, 256, 0, stream>>>(
            lin_r_w + (size_t)i * HID * HID, pR[i], HID, HID);
    }
    pack_weights_kernel<<<(matFrags + 255) / 256, 256, 0, stream>>>(lin1_w, pLin1, HID, HID);

    // ---- degrees -> inv_deg ----
    hipMemsetAsync(invdeg, 0, (size_t)N_NODES * sizeof(float), stream);
    degree_kernel<<<N_EDGES / 256, 256, 0, stream>>>(dst, invdeg, N_EDGES);
    invert_deg_kernel<<<N_NODES / 256, 256, 0, stream>>>(invdeg, N_NODES);

    // ---- embed: hA = x @ embed_w + embed_b ----
    gemm_wmma_kernel<<<N_NODES / 64, 256, 0, stream>>>(
        x, nullptr, pEmbed, nullptr, nullptr, embed_b, hA, N_NODES, IN_CH, 0);

    // ---- 3 SAGE layers (ping-pong hA <-> hB) ----
    float* hin  = hA;
    float* hout = hB;
    for (int i = 0; i < 3; ++i) {
        hipMemsetAsync(agg, 0, szH, stream);
        scatter_add_kernel<<<N_EDGES / 8, 256, 0, stream>>>(src, dst, hin, agg, N_EDGES);
        // hout = (agg * inv_deg) @ lin_l + lin_l_b + hin @ lin_r  [+ LeakyReLU]
        gemm_wmma_kernel<<<N_NODES / 64, 256, 0, stream>>>(
            agg, invdeg, pL[i], hin, pR[i], lin_l_b + (size_t)i * HID,
            hout, N_NODES, HID, (i < 2) ? 1 : 0);
        float* t = hin; hin = hout; hout = t;
    }
    // final node features now in `hin`

    // ---- global add pool ----
    hipMemsetAsync(pooled, 0, (size_t)N_GRAPHS * HID * sizeof(float), stream);
    pool_kernel<<<N_NODES / 8, 256, 0, stream>>>(batch, hin, pooled, N_NODES);

    // ---- MLP head ----
    gemm_wmma_kernel<<<N_GRAPHS / 64, 256, 0, stream>>>(
        pooled, nullptr, pLin1, nullptr, nullptr, lin1_b, hidden, N_GRAPHS, HID, 1);
    out_kernel<<<N_GRAPHS / 8, 256, 0, stream>>>(hidden, lin2_w, lin2_b, out, N_GRAPHS);
}
